// SpikeToCalciumDoubleExp_68891275428038
// MI455X (gfx1250) — matrix operand order repaired and checked by reading, not compiled, not run on hardware
//
#include <hip/hip_runtime.h>

// Spike-to-calcium double-exp FIR (K=121) as Toeplitz-banded WMMA GEMM.
// Out(16x16) = W(16x160) x U(160x16): M = time offset, N = batch row,
// K = padded input window. Split-f16 (hi+lo) 3-term WMMA for ~f32 accuracy
// while staying on the fp16 matrix pipe -> memory-bound (~10.6us floor).
// v2: branch-free A gather via zero-padded LDS w_ext; 3 independent
// accumulators to break the WMMA RAW chain.

typedef _Float16 v16h __attribute__((ext_vector_type(16)));
typedef _Float16 v8h  __attribute__((ext_vector_type(8)));
typedef _Float16 v4h  __attribute__((ext_vector_type(4)));
typedef float    v8f  __attribute__((ext_vector_type(8)));

#define T_OUT   30000
#define T_IN    30120
#define KLEN    121
#define ROWS    1024
#define TCHUNK  512            // outputs per workgroup
#define TILES_PER_WG   32
#define TILES_PER_WAVE 4
#define N_TILES 1875           // 30000 / 16
#define KWIN    160            // padded K window (>= 121+15=136, mult of 32)
#define NKC     5              // K chunks of 32
#define STAGE_LEN 656          // (TILES_PER_WG-1)*16 + KWIN
#define STAGE_V4  (STAGE_LEN/4)
#define LDS_STRIDE 680         // row stride in halves: 680*2B=1360B, 16 rows hit
                               // all 64 LDS banks exactly once for b128 loads
#define WEXT    176            // zero-padded reversed kernel, idx in [0,174]

__global__ __launch_bounds__(256)
void SpikeToCalcium_conv_wmma(const float* __restrict__ u,
                              const float* __restrict__ kern,
                              float* __restrict__ out)
{
    __shared__ _Float16 lds_hi[16 * LDS_STRIDE];
    __shared__ _Float16 lds_lo[16 * LDS_STRIDE];
    __shared__ _Float16 lds_whi[WEXT];
    __shared__ _Float16 lds_wlo[WEXT];

    const int tid   = threadIdx.x;
    const int chunk = blockIdx.x;
    const int rowb  = blockIdx.y * 16;
    const int t0    = chunk * TCHUNK;

    const int lane = tid & 31;
    const int m    = lane & 15;   // A: row M ; B: col N ; same lane slot
    const int h    = lane >> 4;   // lane half

    // ---- stage zero-padded reversed kernel (hi/lo split) into LDS --------
    // w_ext[t] = kern[135 - t] for t in [15,135], else 0.  (t = d + 15)
    if (tid < WEXT) {
        float wv = (tid >= 15 && tid <= 135) ? kern[135 - tid] : 0.0f;
        _Float16 whi = (_Float16)wv;
        lds_whi[tid] = whi;
        lds_wlo[tid] = (_Float16)(wv - (float)whi);
    }

    // ---- stage input window into LDS as split f16 (hi + lo) --------------
    for (int it = 0;; ++it) {
        int idx = tid + it * 256;
        if (idx >= 16 * STAGE_V4) break;
        int row  = idx / STAGE_V4;
        int col  = (idx - row * STAGE_V4) * 4;
        int colc = col;
        if (t0 + colc > T_IN - 4) colc = (T_IN - 4) - t0;  // clamp tail reads
        float4 x = *(const float4*)(u + (size_t)(rowb + row) * T_IN + t0 + colc);
        float xs[4] = {x.x, x.y, x.z, x.w};
        v4h hi, lo;
#pragma unroll
        for (int q = 0; q < 4; ++q) {
            _Float16 hh = (_Float16)xs[q];
            hi[q] = hh;
            lo[q] = (_Float16)(xs[q] - (float)hh);
        }
        *(v4h*)(&lds_hi[row * LDS_STRIDE + col]) = hi;
        *(v4h*)(&lds_lo[row * LDS_STRIDE + col]) = lo;
    }
    __syncthreads();

    // ---- gather constant Toeplitz A fragments (branch-free, in-range) ----
    // A[m][k] = w_ext[15 + k - m]; band zeros come from the padding.
    v16h a_hi[NKC], a_lo[NKC];
#pragma unroll
    for (int kc = 0; kc < NKC; ++kc) {
#pragma unroll
        for (int e = 0; e < 16; ++e) {
            // A k order used here (contiguous): lane half h holds K = 16h+0..15 (matches B)
            int k   = 16 * h + e;
            int idx = 15 + 32 * kc + k - m;     // always in [0, 174]
            a_hi[kc][e] = lds_whi[idx];
            a_lo[kc][e] = lds_wlo[idx];
        }
    }

    // ---- WMMA compute: each wave owns 4 consecutive 16-wide output tiles --
    const int wave = tid >> 5;
    for (int tl = 0; tl < TILES_PER_WAVE; ++tl) {
        int tloc = wave * TILES_PER_WAVE + tl;       // 0..31 within chunk
        int gt   = chunk * TILES_PER_WG + tloc;      // global tile
        if (gt >= N_TILES) break;                    // wave-uniform
        int tw = tloc * 16;

        // three independent accumulator chains (one per split term)
        v8f acc0 = {0.f, 0.f, 0.f, 0.f, 0.f, 0.f, 0.f, 0.f};
        v8f acc1 = acc0, acc2 = acc0;

        // B k order used here (contiguous): lanes 0-15 hold K=0..15, lanes 16-31 K=16..31,
        // packed 2/VGPR -> 16 consecutive halves per lane = 2x ds_load_b128.
        const _Float16* bh_base = &lds_hi[m * LDS_STRIDE + tw + 16 * h];
        const _Float16* bl_base = &lds_lo[m * LDS_STRIDE + tw + 16 * h];

#pragma unroll
        for (int kc = 0; kc < NKC; ++kc) {
            v16h bh, bl;
            ((v8h*)&bh)[0] = *(const v8h*)(bh_base + 32 * kc);
            ((v8h*)&bh)[1] = *(const v8h*)(bh_base + 32 * kc + 8);
            ((v8h*)&bl)[0] = *(const v8h*)(bl_base + 32 * kc);
            ((v8h*)&bl)[1] = *(const v8h*)(bl_base + 32 * kc + 8);

            acc0 = __builtin_amdgcn_wmma_f32_16x16x32_f16(
                false, a_hi[kc], false, bh, (short)0, acc0, false, false);
            acc1 = __builtin_amdgcn_wmma_f32_16x16x32_f16(
                false, a_hi[kc], false, bl, (short)0, acc1, false, false);
            acc2 = __builtin_amdgcn_wmma_f32_16x16x32_f16(
                false, a_lo[kc], false, bh, (short)0, acc2, false, false);
        }

        v8f acc = acc0 + (acc1 + acc2);

        // D layout: lane holds M = 8h..8h+7 (contiguous time), N = m (row).
        float* orow = out + (size_t)(rowb + m) * T_OUT + gt * 16 + 8 * h;
        *(float4*)(orow)     = make_float4(acc[0], acc[1], acc[2], acc[3]);
        *(float4*)(orow + 4) = make_float4(acc[4], acc[5], acc[6], acc[7]);
    }
}

extern "C" void kernel_launch(void* const* d_in, const int* in_sizes, int n_in,
                              void* d_out, int out_size, void* d_ws, size_t ws_size,
                              hipStream_t stream) {
    (void)in_sizes; (void)n_in; (void)d_ws; (void)ws_size; (void)out_size;
    const float* u    = (const float*)d_in[0];   // [1024][30120] f32
    const float* kern = (const float*)d_in[1];   // [121] f32
    float* out        = (float*)d_out;           // [1024][30000] f32

    dim3 grid((N_TILES + TILES_PER_WG - 1) / TILES_PER_WG, ROWS / 16); // 59 x 64
    dim3 block(256);
    SpikeToCalcium_conv_wmma<<<grid, block, 0, stream>>>(u, kern, out);
}
